// LSTM_21139829031696
// MI455X (gfx1250) — matrix operand order repaired and checked by reading, not compiled, not run on hardware
//
#include <hip/hip_runtime.h>

// ---------------------------------------------------------------------------
// Problem constants
// ---------------------------------------------------------------------------
#define SEQ_T 500
#define BATCH 512
#define HID   128
#define G4    512          // 4*HID gate width
#define CH    40           // packed-B chunk size in halves (32 data + 8 pad)

typedef __attribute__((ext_vector_type(16))) _Float16 v16h;
typedef __attribute__((ext_vector_type(8)))  _Float16 v8h;
typedef __attribute__((ext_vector_type(8)))  float    v8f;
typedef __attribute__((ext_vector_type(4)))  unsigned int v4u;
typedef __attribute__((ext_vector_type(4)))  int      v4i;
typedef __attribute__((ext_vector_type(8)))  int      v8i;

#if defined(__has_builtin)
#if __has_builtin(__builtin_amdgcn_tensor_load_to_lds) && __has_builtin(__builtin_amdgcn_s_wait_tensorcnt)
#define HAVE_TDM 1
#endif
#endif
#ifndef HAVE_TDM
#define HAVE_TDM 0
#endif

__device__ __forceinline__ v16h ld_pair(const _Float16* p0, const _Float16* p1) {
    union { v16h v; v8h h[2]; } u;
    u.h[0] = *(const v8h*)p0;
    u.h[1] = *(const v8h*)p1;
    return u.v;
}

#if HAVE_TDM
// Issue one TDM 2-D tile load: 16 rows x 512 f32 from gsrc (row stride 512)
// into contiguous LDS at byte offset lds_off.  Wave-level op, EXEC ignored.
// 6-arg builtin form (clang-23 / therock-10.0 headers).
__device__ __forceinline__ void tdm_load_tile(const float* gsrc, unsigned lds_off) {
    unsigned long long ga = (unsigned long long)gsrc;
    v4u g0;
    g0[0] = 1u;                                   // count=1 (valid user descriptor)
    g0[1] = lds_off;                              // lds_addr (bytes)
    g0[2] = (unsigned)(ga & 0xFFFFFFFFu);         // global_addr[31:0]
    g0[3] = (unsigned)((ga >> 32) & 0x01FFFFFFu)  // global_addr[56:32]
          | (2u << 30);                           // type=2 ("image")
    v8i g1;
    g1[0] = (int)(2u << 16);                      // workgroup_mask=0, data_size=4B
    g1[1] = (int)(512u << 16);                    // tensor_dim0 = 512 (bits 79:48)
    g1[2] = (int)(16u << 16);                     // tensor_dim1 = 16  (bits 111:80)
    g1[3] = (int)(512u << 16);                    // tile_dim0 = 512   (bits 127:112)
    g1[4] = 16;                                   // tile_dim1 = 16    (bits 143:128)
    g1[5] = 512;                                  // tensor_dim0_stride = 512 (bits 207:160)
    g1[6] = 0;
    g1[7] = 0;
    v4i gz4 = (v4i){0, 0, 0, 0};
    v8i gz8 = (v8i){0, 0, 0, 0, 0, 0, 0, 0};
    __builtin_amdgcn_tensor_load_to_lds(g0, g1, gz4, gz4, gz8, 0);
}
#endif

// ---------------------------------------------------------------------------
// Pack a weight matrix w[G4][K] (fp32, PyTorch layout: out = in @ w^T) into the
// WMMA B-operand layout, f16:
//   out[(kt*G4 + n)*CH + h*16 + i] = w[n][kt*32 + h*16 + i]   (zero-padded K)
// ---------------------------------------------------------------------------
__global__ void pack_b_kernel(const float* __restrict__ w, _Float16* __restrict__ out,
                              int K, int nkt) {
    int idx = blockIdx.x * 256 + threadIdx.x;
    int total = nkt * G4 * CH;
    if (idx >= total) return;
    int c  = idx % CH;
    int n  = (idx / CH) % G4;
    int kt = idx / (CH * G4);
    _Float16 v = (_Float16)0.0f;
    if (c < 32) {
        int h = c >> 4;
        int i = c & 15;
        int k = kt * 32 + h * 16 + i;
        if (k < K) v = (_Float16)w[(size_t)n * K + k];
    }
    out[idx] = v;
}

// ---------------------------------------------------------------------------
// Input-projection GEMM:  out[r][n] = bias[n] + sum_k A[r][k] * w^T[k][n]
//   mode 0: A row r=(t*B+b) taken from x[b][t][0:52]         (K=52,  nkt=2)
//   mode 1: A row r=(t*B+b) = concat(hs_f[t][b], hs_r[t][b]) (K=256, nkt=8)
// Per WG: 32 rows x 512 cols, 256 threads (8 waves x 4 col-tiles x 2 row-tiles)
// ---------------------------------------------------------------------------
__global__ void proj_kernel(const float* __restrict__ x,
                            const float* __restrict__ hsf,
                            const float* __restrict__ hsr,
                            const _Float16* __restrict__ wpk,
                            const float* __restrict__ bias,
                            float* __restrict__ out,
                            int K, int nkt, int aStride, int mode) {
    extern __shared__ char smem[];
    _Float16* sA = (_Float16*)smem;                    // [32][aStride]
    const int tid = threadIdx.x;
    const int r0 = blockIdx.x * 32;
    const int Kpad = nkt * 32;

    for (int e = tid; e < 32 * Kpad; e += 256) {
        int row = e / Kpad, k = e % Kpad;
        int r = r0 + row;
        int t = r >> 9;            // /BATCH
        int b = r & (BATCH - 1);
        float v = 0.0f;
        if (k < K) {
            if (mode == 0)
                v = x[(size_t)b * (SEQ_T * 52) + (size_t)t * 52 + k];
            else
                v = (k < HID) ? hsf[((size_t)t * BATCH + b) * HID + k]
                              : hsr[((size_t)t * BATCH + b) * HID + (k - HID)];
        }
        sA[row * aStride + k] = (_Float16)v;
    }
    __syncthreads();

    const int lane = tid & 31, wid = tid >> 5;
    const int l16 = lane & 15,  hi = lane >> 4;
    const int n0w = wid * 64;

    v8f acc[2][4];
    for (int mt = 0; mt < 2; ++mt)
        for (int nt = 0; nt < 4; ++nt) {
            float bv = bias[n0w + nt * 16 + l16];
            for (int j = 0; j < 8; ++j) acc[mt][nt][j] = bv;
        }

    for (int kt = 0; kt < nkt; ++kt) {
        v16h a[2];
#pragma unroll
        for (int mt = 0; mt < 2; ++mt) {
            const _Float16* ap = sA + (mt * 16 + l16) * aStride + kt * 32 + hi * 16;
            a[mt] = ld_pair(ap, ap + 8);
        }
#pragma unroll
        for (int nt = 0; nt < 4; ++nt) {
            const _Float16* bp = wpk + (size_t)(kt * G4 + n0w + nt * 16 + l16) * CH + hi * 16;
            v16h bm = ld_pair(bp, bp + 8);
            acc[0][nt] = __builtin_amdgcn_wmma_f32_16x16x32_f16(
                false, a[0], false, bm, (short)0, acc[0][nt], false, false);
            acc[1][nt] = __builtin_amdgcn_wmma_f32_16x16x32_f16(
                false, a[1], false, bm, (short)0, acc[1][nt], false, false);
        }
    }

#pragma unroll
    for (int mt = 0; mt < 2; ++mt)
#pragma unroll
        for (int nt = 0; nt < 4; ++nt) {
            int col  = n0w + nt * 16 + l16;
            int rowb = mt * 16 + hi * 8;
#pragma unroll
            for (int j = 0; j < 8; ++j)
                out[(size_t)(r0 + rowb + j) * G4 + col] = acc[mt][nt][j];
        }
}

// ---------------------------------------------------------------------------
// Persistent LSTM recurrence: one WG owns 16 batch rows for all 500 steps.
// W_hh^T (packed f16) lives in LDS (160 KB).  Per-step xproj tiles are
// double-buffered in LDS via the Tensor Data Mover (one tensor_load_to_lds
// per step, prefetched one step ahead, synchronized with s_wait_tensorcnt).
// Gate pre-activations exchanged via LDS; c kept in registers.
// ---------------------------------------------------------------------------
#define SW_BYTES  ((size_t)4 * G4 * CH * 2)          // 163840
#define SG_BYTES  ((size_t)16 * 516 * 4)             // 33024
#define SH_BYTES  ((size_t)16 * 136 * 2)             // 4352
#define SXP_BYTES ((size_t)2 * 16 * G4 * 4)          // 65536 (double buffer)
#if HAVE_TDM
#define REC_SHMEM (SW_BYTES + SG_BYTES + SH_BYTES + SXP_BYTES)
#else
#define REC_SHMEM (SW_BYTES + SG_BYTES + SH_BYTES)
#endif

__device__ __forceinline__ void rec_body(const float* __restrict__ xp,
                                         const _Float16* __restrict__ wpk,
                                         float* __restrict__ hs_out,   // [T,B,HID] or null
                                         float* __restrict__ hlast,    // [B,HID] if hs_out null
                                         int reverse, int b0) {
    extern __shared__ char smem[];
    _Float16* sW = (_Float16*)smem;                                    // 4*G4*CH halves
    float*    sG = (float*)(smem + SW_BYTES);                          // [16][516]
    _Float16* sH = (_Float16*)(smem + SW_BYTES + SG_BYTES);            // [16][136]
#if HAVE_TDM
    float*    sXP = (float*)(smem + SW_BYTES + SG_BYTES + SH_BYTES);   // [2][16*G4]
#endif

    const int tid = threadIdx.x;
    for (int e = tid; e < 4 * G4 * CH; e += 256) sW[e] = wpk[e];
    for (int e = tid; e < 16 * 136; e += 256)    sH[e] = (_Float16)0.0f;

    const int lane = tid & 31, wid = tid >> 5;
    const int l16 = lane & 15,  hi = lane >> 4;
    const int jcol = tid & 127;          // elementwise: fixed hidden column
    const int mrow0 = tid >> 7;          // 0 -> even rows, 1 -> odd rows

#if HAVE_TDM
    const unsigned xpBase = (unsigned)(unsigned long long)(const void*)sXP;
    // Prefetch step 0 tile while W is still being staged.
    if (wid == 0) {
        const int tt0 = reverse ? (SEQ_T - 1) : 0;
        tdm_load_tile(xp + ((size_t)tt0 * BATCH + b0) * G4, xpBase);
    }
#endif
    __syncthreads();

    float c[8];
#pragma unroll
    for (int s = 0; s < 8; ++s) c[s] = 0.0f;

    for (int t = 0; t < SEQ_T; ++t) {
        const int tt = reverse ? (SEQ_T - 1 - t) : t;

#if HAVE_TDM
        // Prefetch next step's tile into the other buffer; overlaps the WMMAs.
        if (wid == 0 && t + 1 < SEQ_T) {
            const int tn = reverse ? (SEQ_T - 2 - t) : (t + 1);
            tdm_load_tile(xp + ((size_t)tn * BATCH + b0) * G4,
                          xpBase + (unsigned)(((t + 1) & 1) * 16 * G4 * 4));
        }
#else
        const float* xpt = xp + ((size_t)tt * BATCH + b0) * G4;
#endif

        v8f acc[4];
#pragma unroll
        for (int nt = 0; nt < 4; ++nt)
#pragma unroll
            for (int j = 0; j < 8; ++j) acc[nt][j] = 0.0f;

#pragma unroll
        for (int kt = 0; kt < 4; ++kt) {
            const _Float16* ap = sH + l16 * 136 + kt * 32 + hi * 16;
            v16h a = ld_pair(ap, ap + 8);
#pragma unroll
            for (int nt = 0; nt < 4; ++nt) {
                const _Float16* bp =
                    sW + (size_t)(kt * G4 + wid * 64 + nt * 16 + l16) * CH + hi * 16;
                v16h bm = ld_pair(bp, bp + 8);
                acc[nt] = __builtin_amdgcn_wmma_f32_16x16x32_f16(
                    false, a, false, bm, (short)0, acc[nt], false, false);
            }
        }

#pragma unroll
        for (int nt = 0; nt < 4; ++nt) {
            int col = wid * 64 + nt * 16 + l16;
            int rb  = hi * 8;
#pragma unroll
            for (int j = 0; j < 8; ++j) sG[(rb + j) * 516 + col] = acc[nt][j];
        }

#if HAVE_TDM
        // Ensure this step's TDM tile has landed (in-order per wave: waiting
        // for <=1 outstanding leaves only the prefetch for t+1 in flight).
        if (wid == 0) {
            if (t == SEQ_T - 1) __builtin_amdgcn_s_wait_tensorcnt(0);
            else                __builtin_amdgcn_s_wait_tensorcnt(1);
        }
#endif
        __syncthreads();

#if HAVE_TDM
        const float* xps = sXP + (size_t)(t & 1) * 16 * G4;
#endif
#pragma unroll
        for (int s = 0; s < 8; ++s) {
            int m = 2 * s + mrow0;
#if HAVE_TDM
            float gi = sG[m * 516 + jcol]           + xps[m * G4 + jcol];
            float gf = sG[m * 516 + HID + jcol]     + xps[m * G4 + HID + jcol];
            float gg = sG[m * 516 + 2 * HID + jcol] + xps[m * G4 + 2 * HID + jcol];
            float go = sG[m * 516 + 3 * HID + jcol] + xps[m * G4 + 3 * HID + jcol];
#else
            float gi = sG[m * 516 + jcol]           + xpt[(size_t)m * G4 + jcol];
            float gf = sG[m * 516 + HID + jcol]     + xpt[(size_t)m * G4 + HID + jcol];
            float gg = sG[m * 516 + 2 * HID + jcol] + xpt[(size_t)m * G4 + 2 * HID + jcol];
            float go = sG[m * 516 + 3 * HID + jcol] + xpt[(size_t)m * G4 + 3 * HID + jcol];
#endif
            float sf = 1.0f / (1.0f + __expf(-gf));
            float si = 1.0f / (1.0f + __expf(-gi));
            float so = 1.0f / (1.0f + __expf(-go));
            float cn = sf * c[s] + si * tanhf(gg);
            float hn = so * tanhf(cn);
            c[s] = cn;
            sH[m * 136 + jcol] = (_Float16)hn;
            if (hs_out)
                hs_out[((size_t)tt * BATCH + b0 + m) * HID + jcol] = hn;
            else if (t == SEQ_T - 1)
                hlast[(size_t)(b0 + m) * HID + jcol] = hn;
        }
        __syncthreads();
    }
}

__global__ void lstm_rec1_kernel(const float* __restrict__ xpf, const float* __restrict__ xpr,
                                 const _Float16* __restrict__ wpkf, const _Float16* __restrict__ wpkr,
                                 float* __restrict__ hsf, float* __restrict__ hsr) {
    const int dir = blockIdx.y;
    rec_body(dir ? xpr : xpf, dir ? wpkr : wpkf, dir ? hsr : hsf, nullptr,
             dir, blockIdx.x * 16);
}

__global__ void lstm_rec2_kernel(const float* __restrict__ xp2,
                                 const _Float16* __restrict__ wpk2,
                                 float* __restrict__ hlast) {
    rec_body(xp2, wpk2, nullptr, hlast, 0, blockIdx.x * 16);
}

// ---------------------------------------------------------------------------
// Head: z = selu(last @ W1^T + b1); logits = z @ W2^T + b2l; softmax.
// ---------------------------------------------------------------------------
__global__ void head_kernel(const float* __restrict__ hlast,
                            const float* __restrict__ W1, const float* __restrict__ b1,
                            const float* __restrict__ W2, const float* __restrict__ b2l,
                            float* __restrict__ out) {
    __shared__ float sh[HID];
    __shared__ float sz[300];
    __shared__ float sl[18];
    const int b = blockIdx.x, tid = threadIdx.x;
    if (tid < HID) sh[tid] = hlast[(size_t)b * HID + tid];
    __syncthreads();
    if (tid < 300) {
        float a = b1[tid];
        for (int k = 0; k < HID; ++k) a += sh[k] * W1[(size_t)tid * HID + k];
        const float scale = 1.0507009873554805f, alpha = 1.6732632423543772f;
        sz[tid] = a > 0.0f ? scale * a : scale * alpha * (__expf(a) - 1.0f);
    }
    __syncthreads();
    if (tid < 18) {
        float a = b2l[tid];
        for (int k = 0; k < 300; ++k) a += sz[k] * W2[(size_t)tid * 300 + k];
        sl[tid] = a;
    }
    __syncthreads();
    if (tid == 0) {
        float mx = sl[0];
        for (int i = 1; i < 18; ++i) mx = fmaxf(mx, sl[i]);
        float e[18], sum = 0.0f;
        for (int i = 0; i < 18; ++i) { e[i] = __expf(sl[i] - mx); sum += e[i]; }
        for (int i = 0; i < 18; ++i) out[(size_t)b * 18 + i] = e[i] / sum;
    }
}

// ---------------------------------------------------------------------------
// Host-side orchestration
// ---------------------------------------------------------------------------
extern "C" void kernel_launch(void* const* d_in, const int* in_sizes, int n_in,
                              void* d_out, int out_size, void* d_ws, size_t ws_size,
                              hipStream_t stream) {
    const float* x    = (const float*)d_in[0];
    const float* wihf = (const float*)d_in[1];
    const float* whhf = (const float*)d_in[2];
    const float* bf   = (const float*)d_in[3];
    const float* wihr = (const float*)d_in[4];
    const float* whhr = (const float*)d_in[5];
    const float* br   = (const float*)d_in[6];
    const float* wih2 = (const float*)d_in[7];
    const float* whh2 = (const float*)d_in[8];
    const float* b2   = (const float*)d_in[9];
    const float* W1   = (const float*)d_in[10];
    const float* b1   = (const float*)d_in[11];
    const float* W2   = (const float*)d_in[12];
    const float* b2l  = (const float*)d_in[13];
    float* out = (float*)d_out;

    char* ws = (char*)d_ws;
    size_t off = 0;
    auto alloc = [&](size_t bytes) -> void* {
        void* p = ws + off;
        off += (bytes + 255) & ~(size_t)255;
        return p;
    };
    const size_t xprojBytes = (size_t)SEQ_T * BATCH * G4 * 4;
    const size_t hsBytes    = (size_t)SEQ_T * BATCH * HID * 4;
    float* xpf   = (float*)alloc(xprojBytes);
    float* xpr   = (float*)alloc(xprojBytes);
    float* xp2   = (float*)alloc(xprojBytes);
    float* hsf   = (float*)alloc(hsBytes);
    float* hsr   = (float*)alloc(hsBytes);
    float* hlast = (float*)alloc((size_t)BATCH * HID * 4);
    _Float16* pk_ihf = (_Float16*)alloc((size_t)2 * G4 * CH * 2);
    _Float16* pk_ihr = (_Float16*)alloc((size_t)2 * G4 * CH * 2);
    _Float16* pk_hhf = (_Float16*)alloc((size_t)4 * G4 * CH * 2);
    _Float16* pk_hhr = (_Float16*)alloc((size_t)4 * G4 * CH * 2);
    _Float16* pk_ih2 = (_Float16*)alloc((size_t)8 * G4 * CH * 2);
    _Float16* pk_hh2 = (_Float16*)alloc((size_t)4 * G4 * CH * 2);

    auto packGrid = [](int nkt) { return (nkt * G4 * CH + 255) / 256; };
    pack_b_kernel<<<packGrid(2), 256, 0, stream>>>(wihf, pk_ihf, 52, 2);
    pack_b_kernel<<<packGrid(2), 256, 0, stream>>>(wihr, pk_ihr, 52, 2);
    pack_b_kernel<<<packGrid(4), 256, 0, stream>>>(whhf, pk_hhf, HID, 4);
    pack_b_kernel<<<packGrid(4), 256, 0, stream>>>(whhr, pk_hhr, HID, 4);
    pack_b_kernel<<<packGrid(8), 256, 0, stream>>>(wih2, pk_ih2, 2 * HID, 8);
    pack_b_kernel<<<packGrid(4), 256, 0, stream>>>(whh2, pk_hh2, HID, 4);

    // Layer-1 input projections (K=52 padded to 64)
    const int projBlocks = (SEQ_T * BATCH) / 32;          // 8000
    const size_t sh1 = (size_t)32 * 72 * 2;
    proj_kernel<<<projBlocks, 256, sh1, stream>>>(x, nullptr, nullptr, pk_ihf, bf, xpf,
                                                  52, 2, 72, 0);
    proj_kernel<<<projBlocks, 256, sh1, stream>>>(x, nullptr, nullptr, pk_ihr, br, xpr,
                                                  52, 2, 72, 0);

    // Layer-1 recurrence, both directions concurrently
    lstm_rec1_kernel<<<dim3(BATCH / 16, 2), 256, REC_SHMEM, stream>>>(xpf, xpr, pk_hhf, pk_hhr,
                                                                      hsf, hsr);

    // Layer-2 input projection (K=256)
    const size_t sh2 = (size_t)32 * 264 * 2;
    proj_kernel<<<projBlocks, 256, sh2, stream>>>(nullptr, hsf, hsr, pk_ih2, b2, xp2,
                                                  2 * HID, 8, 264, 1);

    // Layer-2 recurrence (only final h needed)
    lstm_rec2_kernel<<<dim3(BATCH / 16), 256, REC_SHMEM, stream>>>(xp2, pk_hh2, hlast);

    // MLP head + softmax
    head_kernel<<<BATCH, 320, 0, stream>>>(hlast, W1, b1, W2, b2l, out);
}
